// FFTBlock_29884382445710
// MI455X (gfx1250) — matrix operand order repaired and checked by reading, not compiled, not run on hardware
//
#include <hip/hip_runtime.h>
#include <math.h>

typedef unsigned short u16;
typedef __attribute__((ext_vector_type(16))) __bf16 v16bf;
typedef __attribute__((ext_vector_type(8)))  float  v8f;

union FragAB { v16bf v; uint4 q[2]; };
union FragC  { v8f   v; float f[8]; };

#define HAS_TDM (__has_builtin(__builtin_amdgcn_tensor_load_to_lds) && \
                 __has_builtin(__builtin_amdgcn_s_wait_tensorcnt))

__device__ __forceinline__ u16 f2bf(float f) {
  unsigned u = __float_as_uint(f);
  u += 0x7FFFu + ((u >> 16) & 1u);            // round-to-nearest-even
  return (u16)(u >> 16);
}

// ---------------------------------------------------------------- elementwise
__global__ void k_cvt_bf16(const float* __restrict__ in, u16* __restrict__ out, int n) {
  int i = blockIdx.x * blockDim.x + threadIdx.x;
  if (i < n) out[i] = f2bf(in[i]);
}

__global__ void k_zero_u16(u16* __restrict__ p, int n) {    // n multiple of 8
  int i = blockIdx.x * blockDim.x + threadIdx.x;
  if (i * 8 < n) ((uint4*)p)[i] = make_uint4(0u, 0u, 0u, 0u);
}

// ---------------------------------------------------------------- WMMA GEMM
// C[M,N] = A[M,Kd] * B[Kd,N] (+bias)(+relu); A row-major, B given pre-TRANSPOSED
// as BT[N,Kd] so both operand tiles are K-contiguous.
// A element: A[m*lda + (m/aRows)*aPad + k]  (overlapping rows for conv im2col;
//            64-row tiles never straddle a batch since S % 64 == 0)
// C element: C[m*ldc + (m/cRows)*cPad + n]
#define GEMM_LS 72      // LDS row stride in halfwords (64 data + 8 pad, 16B aligned)

__device__ __forceinline__ void wmma_compute(const u16* __restrict__ As,
                                             const u16* __restrict__ Bs,
                                             FragC acc[][2], int wm, int wn,
                                             int lr, int kh) {
#pragma unroll
  for (int kk = 0; kk < 64; kk += 32) {
    FragAB fa[2], fb[2];
#pragma unroll
    for (int mi = 0; mi < 2; ++mi) {
      // A frag: lane<16 -> K 0..7 & 16..23 ; lane>=16 -> K 8..15 & 24..31
      const u16* p = &As[(wm + (mi << 4) + lr) * GEMM_LS + kk + (kh << 3)];
      fa[mi].q[0] = *(const uint4*)p;
      fa[mi].q[1] = *(const uint4*)(p + 16);
    }
#pragma unroll
    for (int ni = 0; ni < 2; ++ni) {
      // B frag: lane -> column n; lane<16 -> K 0..7 & 16..23 ; lane>=16 -> K 8..15 & 24..31
      const u16* p = &Bs[(wn + (ni << 4) + lr) * GEMM_LS + kk + (kh << 3)];
      fb[ni].q[0] = *(const uint4*)p;
      fb[ni].q[1] = *(const uint4*)(p + 16);
    }
#pragma unroll
    for (int mi = 0; mi < 2; ++mi)
#pragma unroll
      for (int ni = 0; ni < 2; ++ni)
        acc[mi][ni].v = __builtin_amdgcn_wmma_f32_16x16x32_bf16(
            false, fa[mi].v, false, fb[ni].v, (short)0, acc[mi][ni].v, false, false);
  }
}

#if HAS_TDM
typedef unsigned int u32x4 __attribute__((ext_vector_type(4)));
typedef int          i32x4 __attribute__((ext_vector_type(4)));
typedef int          i32x8 __attribute__((ext_vector_type(8)));

// DMA a 64x64 bf16 tile (row stride = strideElems halfwords) into LDS, padding
// 4 dwords after every 32 dwords stored -> LDS row stride GEMM_LS halfwords.
__device__ __forceinline__ void tdm_load_tile(const u16* gsrc, unsigned ldsByte,
                                              unsigned strideElems) {
  const unsigned long long ga = (unsigned long long)(uintptr_t)gsrc;
  u32x4 g0;
  g0[0] = 1u;                                               // count=1, user mode
  g0[1] = ldsByte;                                          // lds_addr
  g0[2] = (unsigned)ga;                                     // global_addr[31:0]
  g0[3] = (unsigned)((ga >> 32) & 0x01FFFFFFu) | (2u << 30);// addr[56:32] | type=2
  i32x8 g1;
  g1[0] = (1 << 16) | (1 << 20) | (4 << 22) | (3 << 25);    // 2B elems, pad_en, 32dw->+4dw
  g1[1] = 0;                                                // tensor_dim0 = 1<<30 (lo16=0)
  g1[2] = 0x4000;                                           // dim0 hi16 ; dim1 lo16=0
  g1[3] = 0x4000 | (64 << 16);                              // dim1 hi16 ; tile_dim0=64
  g1[4] = 64;                                               // tile_dim1=64, tile_dim2=0
  g1[5] = (int)strideElems;                                 // tensor_dim0_stride lo32
  g1[6] = 0;
  g1[7] = 0;
  const i32x4 z4 = {0, 0, 0, 0};
  const i32x8 z8 = {0, 0, 0, 0, 0, 0, 0, 0};
  __builtin_amdgcn_tensor_load_to_lds(g0, g1, z4, z4, z8, 0);  // 6-arg (clang-23) form
}
#else
__device__ __forceinline__ void load_tile(const u16* __restrict__ src, int stride,
                                          int tid, uint4 r[4]) {
#pragma unroll
  for (int it = 0; it < 4; ++it) {
    int idx = tid + (it << 7);
    r[it] = *(const uint4*)(src + (size_t)(idx >> 3) * stride + ((idx & 7) << 3));
  }
}
__device__ __forceinline__ void store_tile(u16* __restrict__ dst, int tid,
                                           const uint4 r[4]) {
#pragma unroll
  for (int it = 0; it < 4; ++it) {
    int idx = tid + (it << 7);
    *(uint4*)(dst + (idx >> 3) * GEMM_LS + ((idx & 7) << 3)) = r[it];
  }
}
#endif

__global__ __launch_bounds__(128) void k_gemm_bf16(
    const u16* __restrict__ A, const u16* __restrict__ BT,
    const float* __restrict__ bias, void* __restrict__ Cout,
    int M, int N, int Kd,
    int lda, int aRows, long aPad, int ldb,
    int ldc, int cRows, long cPad, int flags)   // 1=relu 2=store bf16 4=bias
{
#if HAS_TDM
  __shared__ __attribute__((aligned(16))) u16 As_[2][64 * GEMM_LS];
  __shared__ __attribute__((aligned(16))) u16 Bs_[2][64 * GEMM_LS];
#else
  __shared__ __attribute__((aligned(16))) u16 As_[1][64 * GEMM_LS];
  __shared__ __attribute__((aligned(16))) u16 Bs_[1][64 * GEMM_LS];
#endif
  const int tid  = threadIdx.x;
  const int lane = tid & 31;
  const int wave = tid >> 5;
  const int wm   = (wave >> 1) << 5;
  const int wn   = (wave &  1) << 5;
  const int mBase = blockIdx.y << 6;
  const int nBase = blockIdx.x << 6;
  const int lr = lane & 15;
  const int kh = lane >> 4;

  const size_t aBase = (size_t)mBase * lda + (size_t)(mBase / aRows) * aPad;
  const size_t bBase = (size_t)nBase * (size_t)ldb;
  const int nT = Kd >> 6;

  FragC acc[2][2];
#pragma unroll
  for (int mi = 0; mi < 2; ++mi)
#pragma unroll
    for (int ni = 0; ni < 2; ++ni)
#pragma unroll
      for (int j = 0; j < 8; ++j) acc[mi][ni].f[j] = 0.0f;

#if HAS_TDM
  const unsigned ldsA0 = (unsigned)(uintptr_t)&As_[0][0];
  const unsigned ldsB0 = (unsigned)(uintptr_t)&Bs_[0][0];
  const unsigned bufBytesA = (unsigned)sizeof(As_[0]);
  const unsigned bufBytesB = (unsigned)sizeof(Bs_[0]);
  if (tid == 0) {                       // tile 0 -> buffer 0 (TDM, ASYNC DMA)
    tdm_load_tile(A + aBase, ldsA0, (unsigned)lda);
    tdm_load_tile(BT + bBase, ldsB0, (unsigned)ldb);
  }
  for (int i = 0; i < nT; ++i) {
    const int buf = i & 1;
    if (i + 1 < nT) {                   // prefetch tile i+1 into other buffer
      if (tid == 0) {
        const size_t ko = (size_t)(i + 1) << 6;
        tdm_load_tile(A + aBase + ko, ldsA0 + (buf ^ 1) * bufBytesA, (unsigned)lda);
        tdm_load_tile(BT + bBase + ko, ldsB0 + (buf ^ 1) * bufBytesB, (unsigned)ldb);
      }
      __builtin_amdgcn_s_wait_tensorcnt(2);   // in-order: tile i's 2 DMAs done
    } else {
      __builtin_amdgcn_s_wait_tensorcnt(0);
    }
    __syncthreads();                    // publish tile i to all waves
    wmma_compute(&As_[buf][0], &Bs_[buf][0], acc, wm, wn, lr, kh);
    __syncthreads();                    // all waves done with buf before reuse
  }
#else
  uint4 ra[4], rb[4];                   // register-pipelined staging
  load_tile(A + aBase, lda, tid, ra);
  load_tile(BT + bBase, ldb, tid, rb);
  for (int i = 0; i < nT; ++i) {
    store_tile(&As_[0][0], tid, ra);
    store_tile(&Bs_[0][0], tid, rb);
    __syncthreads();
    if (i + 1 < nT) {                   // overlap next tile's loads with compute
      const size_t ko = (size_t)(i + 1) << 6;
      load_tile(A + aBase + ko, lda, tid, ra);
      load_tile(BT + bBase + ko, ldb, tid, rb);
    }
    wmma_compute(&As_[0][0], &Bs_[0][0], acc, wm, wn, lr, kh);
    __syncthreads();
  }
#endif

  const bool relu = flags & 1;
  const bool sbf  = flags & 2;
  const bool hb   = flags & 4;
  const size_t cOff = (size_t)(mBase / cRows) * cPad;     // tile-uniform remap
#pragma unroll
  for (int mi = 0; mi < 2; ++mi) {
#pragma unroll
    for (int ni = 0; ni < 2; ++ni) {
      const int nG = nBase + wn + (ni << 4) + lr;
      const float bval = hb ? bias[nG] : 0.0f;
#pragma unroll
      for (int r = 0; r < 8; ++r) {
        const int mG = mBase + wm + (mi << 4) + (kh << 3) + r;
        float vv = acc[mi][ni].f[r] + bval;
        if (relu) vv = fmaxf(vv, 0.0f);
        const size_t o = (size_t)mG * ldc + cOff + nG;
        if (sbf) ((u16*)Cout)[o] = f2bf(vv);
        else     ((float*)Cout)[o] = vv;
      }
    }
  }
}

// ---------------------------------------------------------------- transpose bf16 [Z,R,C] -> [Z,C,R]
__global__ __launch_bounds__(256) void k_tr_bf16(const u16* __restrict__ in,
                                                 u16* __restrict__ out, int R, int C) {
  __shared__ u16 t[32][33];
  const size_t z = blockIdx.z;
  in  += z * (size_t)R * C;
  out += z * (size_t)R * C;
  const int c0 = blockIdx.x << 5, r0 = blockIdx.y << 5;
  const int tx = threadIdx.x, ty = threadIdx.y;   // block (32,8)
#pragma unroll
  for (int j = 0; j < 4; ++j)
    t[ty + j * 8][tx] = in[(size_t)(r0 + ty + j * 8) * C + c0 + tx];
  __syncthreads();
#pragma unroll
  for (int j = 0; j < 4; ++j)
    out[(size_t)(c0 + ty + j * 8) * R + r0 + tx] = t[tx][ty + j * 8];
}

// ------------------------------------------------- fused convert+transpose f32 [Z,R,C] -> bf16 [Z,C,R]
__global__ __launch_bounds__(256) void k_cvt_tr_bf16(const float* __restrict__ in,
                                                     u16* __restrict__ out, int R, int C) {
  __shared__ u16 t[32][33];
  const size_t z = blockIdx.z;
  in  += z * (size_t)R * C;
  out += z * (size_t)R * C;
  const int c0 = blockIdx.x << 5, r0 = blockIdx.y << 5;
  const int tx = threadIdx.x, ty = threadIdx.y;   // block (32,8)
#pragma unroll
  for (int j = 0; j < 4; ++j)
    t[ty + j * 8][tx] = f2bf(in[(size_t)(r0 + ty + j * 8) * C + c0 + tx]);
  __syncthreads();
#pragma unroll
  for (int j = 0; j < 4; ++j)
    out[(size_t)(c0 + ty + j * 8) * R + r0 + tx] = t[tx][ty + j * 8];
}

// ---------------------------------------------------------------- masked softmax row (S=1024)
__global__ __launch_bounds__(256) void k_softmax(const float* __restrict__ sc,
                                                 u16* __restrict__ pr,
                                                 const unsigned char* __restrict__ mask,
                                                 int S, float scale) {
  __shared__ float red[256];
  const int row = blockIdx.x;
  const int t = threadIdx.x;
  float v[4];
  float mx = -3.0e38f;
#pragma unroll
  for (int i = 0; i < 4; ++i) {
    int c = t + (i << 8);
    float x = sc[(size_t)row * S + c] * scale;
    if (mask[(size_t)row * S + c]) x = -1.0e9f;
    v[i] = x;
    mx = fmaxf(mx, x);
  }
  red[t] = mx; __syncthreads();
  for (int o = 128; o > 0; o >>= 1) { if (t < o) red[t] = fmaxf(red[t], red[t + o]); __syncthreads(); }
  mx = red[0]; __syncthreads();
  float sum = 0.0f;
#pragma unroll
  for (int i = 0; i < 4; ++i) { v[i] = expf(v[i] - mx); sum += v[i]; }
  red[t] = sum; __syncthreads();
  for (int o = 128; o > 0; o >>= 1) { if (t < o) red[t] += red[t + o]; __syncthreads(); }
  const float inv = 1.0f / red[0];
#pragma unroll
  for (int i = 0; i < 4; ++i) pr[(size_t)row * S + t + (i << 8)] = f2bf(v[i] * inv);
}

// ---------------------------------------------------------------- fused add + layernorm (D=384)
__global__ __launch_bounds__(128) void k_add_ln(
    const float* __restrict__ a, const float* __restrict__ b,
    const float* __restrict__ g, const float* __restrict__ be,
    float* __restrict__ outF, u16* __restrict__ outB, int Dm, int Srow) {
  __shared__ float s1[128], s2[128];
  const int row = blockIdx.x;
  const int t = threadIdx.x;
  float v[3];
  float sum = 0.0f, sq = 0.0f;
#pragma unroll
  for (int i = 0; i < 3; ++i) {
    int d = t + (i << 7);
    float x = a[(size_t)row * Dm + d] + b[(size_t)row * Dm + d];
    v[i] = x; sum += x; sq += x * x;
  }
  s1[t] = sum; s2[t] = sq; __syncthreads();
  for (int o = 64; o > 0; o >>= 1) {
    if (t < o) { s1[t] += s1[t + o]; s2[t] += s2[t + o]; }
    __syncthreads();
  }
  const float mean = s1[0] / (float)Dm;
  const float var  = s2[0] / (float)Dm - mean * mean;
  const float inv  = rsqrtf(var + 1e-5f);
  const size_t pb = outB ? ((size_t)row * Dm + (size_t)(row / Srow) * (8 * Dm) + 4 * Dm) : 0;
#pragma unroll
  for (int i = 0; i < 3; ++i) {
    int d = t + (i << 7);
    float y = (v[i] - mean) * inv * g[d] + be[d];
    outF[(size_t)row * Dm + d] = y;
    if (outB) outB[pb + d] = f2bf(y);   // into zero-padded conv input
  }
}

// ---------------------------------------------------------------- host
extern "C" void kernel_launch(void* const* d_in, const int* in_sizes, int n_in,
                              void* d_out, int out_size, void* d_ws, size_t ws_size,
                              hipStream_t stream) {
  (void)in_sizes; (void)n_in; (void)out_size; (void)ws_size;
  const int Bc = 16, Sc = 1024, Dc = 384, Hc = 2, DHc = 192, Fc = 1536, Kc = 9;
  const int Mall = Bc * Sc;

  const float* x   = (const float*)d_in[0];
  const unsigned char* mask = (const unsigned char*)d_in[1];
  const float* Wq  = (const float*)d_in[2];  const float* bq  = (const float*)d_in[3];
  const float* Wk  = (const float*)d_in[4];  const float* bk  = (const float*)d_in[5];
  const float* Wv  = (const float*)d_in[6];  const float* bv  = (const float*)d_in[7];
  const float* Wo  = (const float*)d_in[8];  const float* bo  = (const float*)d_in[9];
  const float* Wc1 = (const float*)d_in[10]; const float* bc1 = (const float*)d_in[11];
  const float* Wc2 = (const float*)d_in[12]; const float* bc2 = (const float*)d_in[13];
  const float* g1  = (const float*)d_in[14]; const float* be1 = (const float*)d_in[15];
  const float* g2  = (const float*)d_in[16]; const float* be2 = (const float*)d_in[17];
  float* out = (float*)d_out;

  char* w = (char*)d_ws; size_t off = 0;
  auto take = [&](size_t bytes) -> char* {
    char* p = w + off; off = (off + bytes + 255) & ~(size_t)255; return p;
  };
  u16*   xbf  = (u16*)  take((size_t)Mall * Dc * 2);
  u16*   WqT  = (u16*)  take((size_t)Hc * Dc * DHc * 2);   // [H][DH][D]
  u16*   WkT  = (u16*)  take((size_t)Hc * Dc * DHc * 2);
  u16*   WvT  = (u16*)  take((size_t)Hc * Dc * DHc * 2);
  u16*   WoT  = (u16*)  take((size_t)Dc * Dc * 2);         // [D][D]
  u16*   Wc1T = (u16*)  take((size_t)Kc * Dc * Fc * 2);    // [F][9D]
  u16*   Wc2T = (u16*)  take((size_t)Kc * Fc * Dc * 2);    // [D][9F]
  u16*   Qb   = (u16*)  take((size_t)Hc * Mall * DHc * 2); // [H][B*S][DH]
  u16*   Kb   = (u16*)  take((size_t)Hc * Mall * DHc * 2); // [H][B*S][DH] (== BT of QK^T)
  u16*   Vb   = (u16*)  take((size_t)Hc * Mall * DHc * 2);
  u16*   Vtr  = (u16*)  take((size_t)Hc * Mall * DHc * 2); // [H*B][DH][S]
  float* scb  = (float*)take((size_t)Sc * Sc * 4);
  u16*   prb  = (u16*)  take((size_t)Sc * Sc * 2);
  u16*   Obf  = (u16*)  take((size_t)Mall * Dc * 2);
  float* attnC= (float*)take((size_t)Mall * Dc * 4);
  float* res2 = (float*)take((size_t)Mall * Dc * 4);
  u16*   xpad = (u16*)  take((size_t)Bc * (Sc + 8) * Dc * 2);
  u16*   hpad = (u16*)  take((size_t)Bc * (Sc + 8) * Fc * 2);
  float* c2o  = (float*)take((size_t)Mall * Dc * 4);

  // ---- operand preparation: x -> bf16, every weight -> transposed bf16
  k_cvt_bf16<<<dim3((unsigned)(((size_t)Mall * Dc + 255) / 256)), 256, 0, stream>>>(
      x, xbf, Mall * Dc);
  k_cvt_tr_bf16<<<dim3(DHc / 32, Dc / 32, Hc), dim3(32, 8), 0, stream>>>(Wq, WqT, Dc, DHc);
  k_cvt_tr_bf16<<<dim3(DHc / 32, Dc / 32, Hc), dim3(32, 8), 0, stream>>>(Wk, WkT, Dc, DHc);
  k_cvt_tr_bf16<<<dim3(DHc / 32, Dc / 32, Hc), dim3(32, 8), 0, stream>>>(Wv, WvT, Dc, DHc);
  k_cvt_tr_bf16<<<dim3(Dc / 32, Dc / 32, 1), dim3(32, 8), 0, stream>>>(Wo, WoT, Dc, Dc);
  k_cvt_tr_bf16<<<dim3(Fc / 32, (Kc * Dc) / 32, 1), dim3(32, 8), 0, stream>>>(Wc1, Wc1T, Kc * Dc, Fc);
  k_cvt_tr_bf16<<<dim3(Dc / 32, (Kc * Fc) / 32, 1), dim3(32, 8), 0, stream>>>(Wc2, Wc2T, Kc * Fc, Dc);

  auto gemm = [&](const u16* A, const u16* BT, const float* bias, void* C,
                  int M, int N, int Kd, int lda, int aRows, long aPad, int ldb,
                  int ldc, int cRows, long cPad, int flags) {
    k_gemm_bf16<<<dim3(N / 64, M / 64), 128, 0, stream>>>(
        A, BT, bias, C, M, N, Kd, lda, aRows, aPad, ldb, ldc, cRows, cPad, flags);
  };

  // ---- QKV projections (per head), outputs bf16 [H][B*S][DH]
  for (int h = 0; h < Hc; ++h) {
    gemm(xbf, WqT + (size_t)h * Dc * DHc, bq + h * DHc, Qb + (size_t)h * Mall * DHc,
         Mall, DHc, Dc, Dc, Mall, 0, Dc, DHc, Mall, 0, 2 | 4);
    gemm(xbf, WkT + (size_t)h * Dc * DHc, bk + h * DHc, Kb + (size_t)h * Mall * DHc,
         Mall, DHc, Dc, Dc, Mall, 0, Dc, DHc, Mall, 0, 2 | 4);
    gemm(xbf, WvT + (size_t)h * Dc * DHc, bv + h * DHc, Vb + (size_t)h * Mall * DHc,
         Mall, DHc, Dc, Dc, Mall, 0, Dc, DHc, Mall, 0, 2 | 4);
  }
  // V -> V^T per (h,b): [S,DH] -> [DH,S]  (B operand of P*V)
  k_tr_bf16<<<dim3(DHc / 32, Sc / 32, Hc * Bc), dim3(32, 8), 0, stream>>>(Vb, Vtr, Sc, DHc);

  // ---- attention per (b,h); K itself is the pre-transposed B of Q*K^T
  const float scale = 1.0f / sqrtf((float)Dc);   // reference scales by sqrt(d_model)
  for (int b = 0; b < Bc; ++b)
    for (int h = 0; h < Hc; ++h) {
      const size_t zi = (size_t)h * Bc + b;
      gemm(Qb + zi * Sc * DHc, Kb + zi * Sc * DHc, nullptr, scb,
           Sc, Sc, DHc, DHc, Sc, 0, DHc, Sc, Sc, 0, 0);
      k_softmax<<<Sc, 256, 0, stream>>>(scb, prb, mask + (size_t)b * Sc * Sc, Sc, scale);
      gemm(prb, Vtr + zi * Sc * DHc, nullptr, Obf + (size_t)b * Sc * Dc + (size_t)h * DHc,
           Sc, DHc, Sc, Sc, Sc, 0, Sc, Dc, Sc, 0, 2);
    }

  // ---- output projection + LN1 (also emits padded bf16 conv input)
  gemm(Obf, WoT, bo, attnC, Mall, Dc, Dc, Dc, Mall, 0, Dc, Dc, Mall, 0, 4);
  {
    int nx = Bc * (Sc + 8) * Dc, nh = Bc * (Sc + 8) * Fc;
    k_zero_u16<<<dim3((unsigned)((nx / 8 + 255) / 256)), 256, 0, stream>>>(xpad, nx);
    k_zero_u16<<<dim3((unsigned)((nh / 8 + 255) / 256)), 256, 0, stream>>>(hpad, nh);
  }
  k_add_ln<<<Mall, 128, 0, stream>>>(attnC, x, g1, be1, res2, xpad, Dc, Sc);

  // ---- conv1 (K=9, D->F) as GEMM over padded rows, fused bias+ReLU, bf16 out into padded h
  gemm(xpad, Wc1T, bc1, hpad + 4 * Fc, Mall, Fc, Kc * Dc,
       Dc, Sc, 8L * Dc, Kc * Dc, Fc, Sc, 8L * Fc, 1 | 2 | 4);
  // ---- conv2 (K=9, F->D) as GEMM, f32 out
  gemm(hpad, Wc2T, bc2, c2o, Mall, Dc, Kc * Fc,
       Fc, Sc, 8L * Fc, Kc * Fc, Dc, Mall, 0, 4);

  // ---- LN2 -> final output
  k_add_ln<<<Mall, 128, 0, stream>>>(c2o, res2, g2, be2, out, nullptr, Dc, Sc);
}
